// COPACRR_5222680232152
// MI455X (gfx1250) — matrix operand order repaired and checked by reading, not compiled, not run on hardware
//
#include <hip/hip_runtime.h>
#include <hip/hip_bf16.h>
#include <math.h>

// ---- problem constants (from reference) ----
#define BB   128
#define QQ   16
#define DD   800
#define EE   300
#define KPAD 320          // K padded to 10 x 32 for wmma_f32_16x16x32_f16
#define NT   (DD / 16)    // 50 doc tiles of N=16
#define WCTX 4
#define TOPK 6
#define KMAX 2
#define SLABW 304         // padded LDS row stride (floats) for the 24-row slab
#define EE4  (EE / 4)     // 75 float4 chunks per embedding row

typedef __attribute__((ext_vector_type(16))) _Float16 v16h;
typedef __attribute__((ext_vector_type(4)))  _Float16 h4;
typedef __attribute__((ext_vector_type(8)))  float    v8f;

// ---------------- workspace layout (bytes) ----------------
// qF16   : B*16*KPAD halves   = 1,310,720
// qnorm  : B*16 f32           = 8,192
// simmat : B*16*800 f32       = 6,553,600
// ctxsim : B*16*800 f32       = 6,553,600
// scores : B*16*13 f32        = 106,496          total ~14.5 MB
#define OFF_QF16   ((size_t)0)
#define OFF_QNORM  ((size_t)1310720)
#define OFF_SIM    ((size_t)1318912)
#define OFF_CTX    ((size_t)7872512)
#define OFF_SCORES ((size_t)14426112)

__device__ __forceinline__ float wave_sum32(float v) {
#pragma unroll
  for (int off = 16; off >= 1; off >>= 1) v += __shfl_xor(v, off, 32);
  return v;
}

// ---------------------------------------------------------------------------
// Kernel 1: gather query embeddings -> f16 (K padded to 320) + norms (+eps)
// grid(B), block(256) = 8 waves; wave w handles rows {w, w+8}
// float4 gathers (rows are 1200 B, 16B-aligned), 8-byte packed f16 stores.
// ---------------------------------------------------------------------------
__global__ void copacrr_qprep(const int* __restrict__ qrls,
                              const float* __restrict__ emb,
                              _Float16* __restrict__ qF,
                              float* __restrict__ qnorm) {
  const int b = blockIdx.x;
  const int lane = threadIdx.x & 31;
  const int wv = threadIdx.x >> 5;
  for (int q = wv; q < QQ; q += 8) {
    const int word = qrls[b * QQ + q];
    const float4* row = (const float4*)(emb + (size_t)word * EE);
    _Float16* orow = qF + (size_t)(b * QQ + q) * KPAD;
    float ss = 0.f;
    for (int c = lane; c < EE4; c += 32) {
      const float4 v = row[c];
      ss += v.x * v.x + v.y * v.y + v.z * v.z + v.w * v.w;
      h4 hv = {(_Float16)v.x, (_Float16)v.y, (_Float16)v.z, (_Float16)v.w};
      *(h4*)(orow + c * 4) = hv;
    }
    for (int e = EE + lane; e < KPAD; e += 32) orow[e] = (_Float16)0.f;
    ss = wave_sum32(ss);
    if (lane == 0) qnorm[b * QQ + q] = sqrtf(ss) + 1e-9f;
  }
}

// ---------------------------------------------------------------------------
// Kernel 2: per (b, doc-tile): gather 24-row fp32 slab (tile + /-4 halo) with
// float4 loads, build f16 demb tile + windowed-mean context tile + norms in
// LDS, then 10 K-steps x 2 WMMAs -> normalized sim & ctxsim 16x16 tiles.
// grid(NT, B), block(32) -- one wave per tile.
// ---------------------------------------------------------------------------
__global__ void copacrr_simtile(const int* __restrict__ dwordsG,
                                const float* __restrict__ emb,
                                const _Float16* __restrict__ qF,
                                const float* __restrict__ qnorm,
                                float* __restrict__ simmat,
                                float* __restrict__ ctxsim) {
  __shared__ float    slab[24 * SLABW];
  __shared__ int      dw[24];
  __shared__ _Float16 dt[16 * KPAD];
  __shared__ _Float16 ct[16 * KPAD];
  __shared__ float    dn[16];
  __shared__ float    cn[16];

  const int tile = blockIdx.x;
  const int b = blockIdx.y;
  const int d0 = tile * 16;
  const int lane = threadIdx.x;

  // --- phase 1: word ids for slab rows, then float4 gather (coalesced) ---
  if (lane < 24) {
    const int g = d0 - WCTX + lane;
    dw[lane] = (g >= 0 && g < DD) ? dwordsG[b * DD + g] : -1;
  }
  __syncthreads();
  for (int idx = lane; idx < 24 * EE4; idx += 32) {
    const int r = idx / EE4;
    const int c = idx - r * EE4;
    const int w = dw[r];
    float4 v = make_float4(0.f, 0.f, 0.f, 0.f);
    if (w >= 0) v = ((const float4*)(emb + (size_t)w * EE))[c];
    *(float4*)&slab[r * SLABW + c * 4] = v;
  }
  // warm L2 for the next tile's 16 fresh rows (global_prefetch_b8)
  if (tile + 1 < NT && lane < 16) {
    const int g = d0 + 16 + WCTX + lane;  // rows not in this slab
    if (g < DD) {
      const int w = dwordsG[b * DD + g];
      __builtin_prefetch(emb + (size_t)w * EE, 0, 0);
    }
  }
  __syncthreads();

  // --- phase 2: f16 demb tile + row norms (2 lanes per row) ---
  {
    const int row = lane >> 1, sub = lane & 1;
    float ss = 0.f;
    for (int e = sub; e < EE; e += 2) {
      float v = slab[(row + WCTX) * SLABW + e];
      ss += v * v;
      dt[row * KPAD + e] = (_Float16)v;
    }
    ss += __shfl_xor(ss, 1, 32);
    if (!sub) dn[row] = sqrtf(ss) + 1e-9f;
  }
  // --- phase 3: context tile = sum over window [i-4, i+4) / 9 ---
  {
    const int row = lane >> 1, sub = lane & 1;
    float ss = 0.f;
    for (int e = sub; e < EE; e += 2) {
      float s = 0.f;
#pragma unroll
      for (int j = 0; j < 8; ++j) s += slab[(row + j) * SLABW + e];
      float v = s * (1.f / 9.f);
      ss += v * v;
      ct[row * KPAD + e] = (_Float16)v;
    }
    ss += __shfl_xor(ss, 1, 32);
    if (!sub) cn[row] = sqrtf(ss) + 1e-9f;
  }
  for (int idx = lane; idx < 16 * (KPAD - EE); idx += 32) {
    const int r = idx / (KPAD - EE);
    const int c = EE + idx % (KPAD - EE);
    dt[r * KPAD + c] = (_Float16)0.f;
    ct[r * KPAD + c] = (_Float16)0.f;
  }
  __syncthreads();

  // --- phase 4: WMMA: A = qemb (16xK), B = tile (Kx16) for sim and ctx ---
  const int m = lane & 15;   // A row / B column
  const int h = lane >> 4;   // lane-half selector per ISA layout
  v8f accS = {};
  v8f accC = {};
  const _Float16* qrowBase = qF + (size_t)(b * QQ + m) * KPAD;
#pragma unroll
  for (int s = 0; s < KPAD / 32; ++s) {
    const int kb = 32 * s;
    v16h a, bd, bc;
    const _Float16* qp = qrowBase + kb + h * 8;
#pragma unroll
    for (int j = 0; j < 8; ++j) {       // A 16-bit 16x32 layout: chunks {0-7},{16-23} (+8 if hi half)
      a[j] = qp[j];
      a[8 + j] = qp[16 + j];
    }
#pragma unroll
    for (int j = 0; j < 16; ++j) {      // B 32x16: lane n holds K chunks {0-7},{16-23} (+8 if hi half) of column n
      bd[j] = dt[m * KPAD + kb + h * 8 + (j & 7) + (j >> 3) * 16];
      bc[j] = ct[m * KPAD + kb + h * 8 + (j & 7) + (j >> 3) * 16];
    }
    accS = __builtin_amdgcn_wmma_f32_16x16x32_f16(false, a, false, bd, (short)0, accS, false, false);
    accC = __builtin_amdgcn_wmma_f32_16x16x32_f16(false, a, false, bc, (short)0, accC, false, false);
  }

  // --- normalize + store: VGPR r -> M = r + 8*h, N = lane&15 ---
  const float dnv = dn[m];
  const float cnv = cn[m];
#pragma unroll
  for (int r = 0; r < 8; ++r) {
    const int mm = r + h * 8;
    const float qv = qnorm[b * QQ + mm];
    const size_t off = ((size_t)(b * QQ + mm)) * DD + d0 + m;
    simmat[off] = accS[r] / (qv * dnv);
    ctxsim[off] = accC[r] / (qv * cnv);
  }
}

// ---------------------------------------------------------------------------
// Kernel 3: per (b,q) wave: 1x1/2x2/3x3 conv + relu + filter-max + top-2,
// top-6 of ctxsim, idf -> scores[b][q][13].
// grid(B), block(512) = 16 waves (one per q).
// ---------------------------------------------------------------------------
__global__ void copacrr_scores(const float* __restrict__ simmat,
                               const float* __restrict__ ctxsim,
                               const int* __restrict__ qrls,
                               const float* __restrict__ idf,
                               const float* __restrict__ c1w, const float* __restrict__ c1b,
                               const float* __restrict__ c2w, const float* __restrict__ c2b,
                               const float* __restrict__ c3w, const float* __restrict__ c3b,
                               float* __restrict__ scores) {
  __shared__ float wsh[544];  // [0,32)=c1w [32,64)=c1b [64,192)=c2w [192,224)=c2b [224,512)=c3w [512,544)=c3b
  const int b = blockIdx.x;
  for (int i = threadIdx.x; i < 544; i += blockDim.x) {
    float v;
    if (i < 32) v = c1w[i];
    else if (i < 64) v = c1b[i - 32];
    else if (i < 192) v = c2w[i - 64];
    else if (i < 224) v = c2b[i - 192];
    else if (i < 512) v = c3w[i - 224];
    else v = c3b[i - 512];
    wsh[i] = v;
  }
  __syncthreads();

  const int q = threadIdx.x >> 5;
  const int lane = threadIdx.x & 31;
  const float* s0 = simmat + (size_t)(b * QQ + q) * DD;
  const bool h1 = (q + 1 < QQ), h2 = (q + 2 < QQ);
  const float* s1 = h1 ? simmat + (size_t)(b * QQ + q + 1) * DD : s0;
  const float* s2 = h2 ? simmat + (size_t)(b * QQ + q + 2) * DD : s0;
  const float* cr = ctxsim + (size_t)(b * QQ + q) * DD;

  float a1[3], a2[3];
#pragma unroll
  for (int k = 0; k < 3; ++k) { a1[k] = -3.0e38f; a2[k] = -3.0e38f; }
  float t6[TOPK];
#pragma unroll
  for (int k = 0; k < TOPK; ++k) t6[k] = -3.0e38f;

  for (int d = lane; d < DD; d += 32) {
    const bool c1ok = (d + 1 < DD), c2ok = (d + 2 < DD);
    const float x00 = s0[d];
    const float x01 = c1ok ? s0[d + 1] : 0.f;
    const float x02 = c2ok ? s0[d + 2] : 0.f;
    const float x10 = h1 ? s1[d] : 0.f;
    const float x11 = (h1 && c1ok) ? s1[d + 1] : 0.f;
    const float x12 = (h1 && c2ok) ? s1[d + 2] : 0.f;
    const float x20 = h2 ? s2[d] : 0.f;
    const float x21 = (h2 && c1ok) ? s2[d + 1] : 0.f;
    const float x22 = (h2 && c2ok) ? s2[d + 2] : 0.f;

    float m1 = 0.f, m2 = 0.f, m3 = 0.f;  // relu outputs are >= 0
    for (int f = 0; f < 32; ++f) {
      const float v1 = fmaxf(0.f, wsh[f] * x00 + wsh[32 + f]);
      const float* w2f = wsh + 64 + f * 4;
      const float v2 = fmaxf(0.f, w2f[0] * x00 + w2f[1] * x01 + w2f[2] * x10 + w2f[3] * x11 + wsh[192 + f]);
      const float* w3f = wsh + 224 + f * 9;
      const float v3 = fmaxf(0.f, w3f[0] * x00 + w3f[1] * x01 + w3f[2] * x02 +
                                  w3f[3] * x10 + w3f[4] * x11 + w3f[5] * x12 +
                                  w3f[6] * x20 + w3f[7] * x21 + w3f[8] * x22 + wsh[512 + f]);
      m1 = fmaxf(m1, v1); m2 = fmaxf(m2, v2); m3 = fmaxf(m3, v3);
    }
    const float mv[3] = {m1, m2, m3};
#pragma unroll
    for (int k = 0; k < 3; ++k) {
      if (mv[k] > a1[k]) { a2[k] = a1[k]; a1[k] = mv[k]; }
      else if (mv[k] > a2[k]) a2[k] = mv[k];
    }
    const float cv = cr[d];
    if (cv > t6[TOPK - 1]) {
      t6[TOPK - 1] = cv;
#pragma unroll
      for (int k = TOPK - 1; k > 0; --k)
        if (t6[k] > t6[k - 1]) { float tmp = t6[k]; t6[k] = t6[k - 1]; t6[k - 1] = tmp; }
    }
  }

  // butterfly merges across the wave32
#pragma unroll
  for (int off = 16; off >= 1; off >>= 1) {
#pragma unroll
    for (int k = 0; k < 3; ++k) {
      const float pa = __shfl_xor(a1[k], off, 32);
      const float pb = __shfl_xor(a2[k], off, 32);
      if (pa > a1[k]) { a2[k] = fmaxf(a1[k], pb); a1[k] = pa; }
      else a2[k] = fmaxf(a2[k], pa);
    }
    float p6[TOPK];
#pragma unroll
    for (int k = 0; k < TOPK; ++k) p6[k] = __shfl_xor(t6[k], off, 32);
#pragma unroll
    for (int k = 0; k < TOPK; ++k) {
      const float cv = p6[k];
      if (cv > t6[TOPK - 1]) {
        t6[TOPK - 1] = cv;
#pragma unroll
        for (int j = TOPK - 1; j > 0; --j)
          if (t6[j] > t6[j - 1]) { float tmp = t6[j]; t6[j] = t6[j - 1]; t6[j - 1] = tmp; }
      }
    }
  }

  if (lane == 0) {
    float* o = scores + (size_t)(b * QQ + q) * 13;
    o[0] = a1[0]; o[1] = a2[0];
    o[2] = a1[1]; o[3] = a2[1];
    o[4] = a1[2]; o[5] = a2[2];
#pragma unroll
    for (int k = 0; k < TOPK; ++k) o[6 + k] = t6[k];
    o[12] = idf[qrls[b * QQ + q]];
  }
}

// ---------------------------------------------------------------------------
// Kernel 4: MLP 208 -> 32 -> 32 -> 1, one wave per batch row.
// ---------------------------------------------------------------------------
__global__ void copacrr_mlp(const float* __restrict__ scores,
                            const float* __restrict__ w1, const float* __restrict__ b1,
                            const float* __restrict__ w2, const float* __restrict__ b2,
                            const float* __restrict__ w3, const float* __restrict__ b3,
                            float* __restrict__ out) {
  __shared__ float xs[QQ * 13];
  __shared__ float h1s[32];
  const int b = blockIdx.x;
  const int lane = threadIdx.x;
  for (int i = lane; i < QQ * 13; i += 32) xs[i] = scores[(size_t)b * QQ * 13 + i];
  __syncthreads();
  float h = b1[lane];
  for (int k = 0; k < QQ * 13; ++k) h += xs[k] * w1[k * 32 + lane];
  h1s[lane] = fmaxf(h, 0.f);
  __syncthreads();
  float g = b2[lane];
#pragma unroll
  for (int k = 0; k < 32; ++k) g += h1s[k] * w2[k * 32 + lane];
  g = fmaxf(g, 0.f);
  float p = wave_sum32(g * w3[lane]);
  if (lane == 0) out[b] = p + b3[0];
}

// ---------------------------------------------------------------------------
extern "C" void kernel_launch(void* const* d_in, const int* in_sizes, int n_in,
                              void* d_out, int out_size, void* d_ws, size_t ws_size,
                              hipStream_t stream) {
  (void)in_sizes; (void)n_in; (void)out_size; (void)ws_size;
  const int*   qrls = (const int*)d_in[0];
  const int*   dwords = (const int*)d_in[1];
  const float* emb = (const float*)d_in[2];
  const float* idf = (const float*)d_in[3];
  const float* c1w = (const float*)d_in[4];
  const float* c1b = (const float*)d_in[5];
  const float* c2w = (const float*)d_in[6];
  const float* c2b = (const float*)d_in[7];
  const float* c3w = (const float*)d_in[8];
  const float* c3b = (const float*)d_in[9];
  const float* w1 = (const float*)d_in[10];
  const float* b1 = (const float*)d_in[11];
  const float* w2 = (const float*)d_in[12];
  const float* b2 = (const float*)d_in[13];
  const float* w3 = (const float*)d_in[14];
  const float* b3 = (const float*)d_in[15];
  float* out = (float*)d_out;

  char* ws = (char*)d_ws;
  _Float16* qF     = (_Float16*)(ws + OFF_QF16);
  float*    qnorm  = (float*)(ws + OFF_QNORM);
  float*    simmat = (float*)(ws + OFF_SIM);
  float*    ctxsim = (float*)(ws + OFF_CTX);
  float*    scores = (float*)(ws + OFF_SCORES);

  copacrr_qprep<<<dim3(BB), dim3(256), 0, stream>>>(qrls, emb, qF, qnorm);
  copacrr_simtile<<<dim3(NT, BB), dim3(32), 0, stream>>>(dwords, emb, qF, qnorm, simmat, ctxsim);
  copacrr_scores<<<dim3(BB), dim3(512), 0, stream>>>(simmat, ctxsim, qrls, idf,
                                                     c1w, c1b, c2w, c2b, c3w, c3b, scores);
  copacrr_mlp<<<dim3(BB), dim3(32), 0, stream>>>(scores, w1, b1, w2, b2, w3, b3, out);
}
